// MyConv2d_64742337020566
// MI455X (gfx1250) — matrix operand order repaired and checked by reading, not compiled, not run on hardware
//
#include <hip/hip_runtime.h>
#include <stdint.h>

// ---------------------------------------------------------------------------
// Implicit-GEMM conv2d for MI455X (gfx1250), bf16x3 split-precision WMMA.
//   A (SRC0) = weight  [M=256 cout][K=2304 = 9 taps * 256 cin]   (bf16 hi/lo)
//   B (SRC1) = im2col(x) [K][N=100352 = 32*56*56]                (bf16 hi/lo)
//   C = f32 accum; D row-major [M][N] == flat [COUT][B][H][W] == d_out layout
// Epilogue uses non-temporal stores so the 103 MB output streams past L2,
// keeping the 103 MB input (re-read ~3.8x by the tiling) L2-resident.
// ---------------------------------------------------------------------------

typedef __attribute__((ext_vector_type(16))) __bf16 v16bf;
typedef __attribute__((ext_vector_type(8)))  float  v8f;

#define CIN   256
#define COUT  256
#define IMHW  3136      // 56*56
#define NTOT  100352    // 32*3136
#define KTOT  2304      // 256*9
#define LROW  20        // LDS dwords per j-row (16 data + 4 pad, 16B aligned)
#define LPLANE (256*LROW)

union Frag {
    uint4 u[2];
    v16bf v;
};

__device__ __forceinline__ uint32_t bf16_rn(float f) {
    uint32_t u = __builtin_bit_cast(uint32_t, f);
    return (u + 0x7FFFu + ((u >> 16) & 1u)) >> 16;
}

// ---- weight prep: [co][ci][3][3] f32 -> hi/lo bf16 planes [co][tap][ci] ----
__global__ void conv_wprep(const float* __restrict__ w,
                           uint16_t* __restrict__ whi,
                           uint16_t* __restrict__ wlo) {
    int idx = blockIdx.x * blockDim.x + threadIdx.x;   // co*2304 + tap*256 + ci
    if (idx >= COUT * KTOT) return;
    int co  = idx / KTOT;
    int rem = idx - co * KTOT;
    int tap = rem >> 8;
    int ci  = rem & 255;
    float f = w[(co * CIN + ci) * 9 + tap];
    uint32_t hi = bf16_rn(f);
    float fhi = __builtin_bit_cast(float, hi << 16);
    uint32_t lo = bf16_rn(f - fhi);
    whi[idx] = (uint16_t)hi;
    wlo[idx] = (uint16_t)lo;
}

// ---- main implicit-GEMM kernel ----
__global__ __launch_bounds__(256) void conv_wmma(
    const float*    __restrict__ x,
    const uint16_t* __restrict__ whi,
    const uint16_t* __restrict__ wlo,
    float*          __restrict__ out)
{
    __shared__ uint32_t lds[2 * LPLANE];   // hi plane, lo plane (40 KB)

    const int tid   = threadIdx.x;
    const int lane  = tid & 31;
    const int wid   = tid >> 5;
    const int wm    = wid & 3;      // 4 waves along M
    const int wn    = wid >> 2;     // 2 waves along N
    const int l15   = lane & 15;
    const int lhalf = lane >> 4;

    const int n0wg  = blockIdx.x * 128;            // N tile (flat b*hw)
    const int mbase = blockIdx.y * 128 + wm * 32;  // wave's M base

    // per-N-subtile column coords (uniform over the whole K loop)
    int hc[4], wc[4], jb[4];
#pragma unroll
    for (int ns = 0; ns < 4; ++ns) {
        int n  = n0wg + wn * 64 + ns * 16 + l15;
        int hw = n % IMHW;
        hc[ns] = hw / 56;
        wc[ns] = hw - hc[ns] * 56;
        jb[ns] = wn * 64 + ns * 16 + l15;
    }

    v8f acc[2][4];
#pragma unroll
    for (int ms = 0; ms < 2; ++ms)
#pragma unroll
        for (int ns = 0; ns < 4; ++ns)
            acc[ms][ns] = {};

    const int kAoff = lhalf ? 8 : 0;   // A element offset within 32-chunk
    const int kBoff = lhalf ? 4 : 0;   // B dword (ci-pair) offset

    for (int c = 0; c < 8; ++c) {          // ci chunks of 32
        const int ci0 = c * 32;

        // ---- stage halo window into LDS as packed bf16-pair hi/lo planes ----
        {
            int j  = tid;                      // 0..255 (242..255 => zero rows)
            int g  = n0wg + j - 57;            // flat pixel index into x
            bool ok = (j < 242) & (g >= 0) & (g < NTOT);
            int gg = ok ? g : 0;
            int b  = gg / IMHW;
            int hw = gg - b * IMHW;
            const float* xp = x + (size_t)(b * CIN + ci0) * IMHW + hw;
#pragma unroll
            for (int i = 0; i < 16; ++i) {
                float f0 = ok ? xp[(size_t)(2 * i) * IMHW] : 0.0f;
                float f1 = ok ? xp[(size_t)(2 * i + 1) * IMHW] : 0.0f;
                uint32_t h0 = bf16_rn(f0), h1 = bf16_rn(f1);
                float fh0 = __builtin_bit_cast(float, h0 << 16);
                float fh1 = __builtin_bit_cast(float, h1 << 16);
                uint32_t l0 = bf16_rn(f0 - fh0), l1 = bf16_rn(f1 - fh1);
                lds[j * LROW + i]          = h0 | (h1 << 16);
                lds[LPLANE + j * LROW + i] = l0 | (l1 << 16);
            }
            // prefetch next ci chunk's window while the matrix pipe works
            if (c < 7) {
#pragma unroll
                for (int i = 0; i < 16; i += 2)
                    __builtin_prefetch(xp + (size_t)(32 + 2 * i) * IMHW, 0, 1);
            }
        }
        __syncthreads();

        // ---- consume: 9 taps x (A loads + 4 N-subtiles x 6 WMMA) ----
#pragma unroll
        for (int tap = 0; tap < 9; ++tap) {
            const int dh = tap / 3, dw = tap % 3;

            Frag Ahi[2], Alo[2];
            const int kelem = tap * 256 + ci0 + kAoff;
#pragma unroll
            for (int ms = 0; ms < 2; ++ms) {
                int mrow = mbase + ms * 16 + l15;
                size_t eoff = (size_t)mrow * KTOT + kelem;
                const uint4* ph = (const uint4*)(whi + eoff);
                const uint4* pl = (const uint4*)(wlo + eoff);
                Ahi[ms].u[0] = ph[0];  Ahi[ms].u[1] = ph[2];  // k..k+7, k+16..k+23
                Alo[ms].u[0] = pl[0];  Alo[ms].u[1] = pl[2];
            }

#pragma unroll
            for (int ns = 0; ns < 4; ++ns) {
                bool valid = ((unsigned)(hc[ns] + dh - 1) < 56u) &
                             ((unsigned)(wc[ns] + dw - 1) < 56u);
                int j   = jb[ns] + dh * 56 + dw;       // <= 241
                int row = valid ? j : 242;             // row 242 is all zeros
                int base = row * LROW + kBoff;

                Frag Bhi, Blo;
                Bhi.u[0] = *(const uint4*)&lds[base];
                Bhi.u[1] = *(const uint4*)&lds[base + 8];
                Blo.u[0] = *(const uint4*)&lds[LPLANE + base];
                Blo.u[1] = *(const uint4*)&lds[LPLANE + base + 8];

#pragma unroll
                for (int ms = 0; ms < 2; ++ms) {
                    acc[ms][ns] = __builtin_amdgcn_wmma_f32_16x16x32_bf16(
                        false, Ahi[ms].v, false, Bhi.v, (short)0, acc[ms][ns], false, false);
                    acc[ms][ns] = __builtin_amdgcn_wmma_f32_16x16x32_bf16(
                        false, Ahi[ms].v, false, Blo.v, (short)0, acc[ms][ns], false, false);
                    acc[ms][ns] = __builtin_amdgcn_wmma_f32_16x16x32_bf16(
                        false, Alo[ms].v, false, Bhi.v, (short)0, acc[ms][ns], false, false);
                }
            }
        }
        __syncthreads();
    }

    // ---- epilogue: C/D layout -> row-major [M][N] == d_out flat layout ----
    // Non-temporal: output is write-once, keep L2 for the input tensor.
#pragma unroll
    for (int ms = 0; ms < 2; ++ms) {
        int mrow0 = mbase + ms * 16 + lhalf * 8;
#pragma unroll
        for (int ns = 0; ns < 4; ++ns) {
            int n = n0wg + wn * 64 + ns * 16 + l15;
#pragma unroll
            for (int r = 0; r < 8; ++r)
                __builtin_nontemporal_store(acc[ms][ns][r],
                                            &out[(size_t)(mrow0 + r) * NTOT + n]);
        }
    }
}

extern "C" void kernel_launch(void* const* d_in, const int* in_sizes, int n_in,
                              void* d_out, int out_size, void* d_ws, size_t ws_size,
                              hipStream_t stream) {
    const float* x = (const float*)d_in[0];
    const float* w = (const float*)d_in[1];
    uint16_t* whi = (uint16_t*)d_ws;
    uint16_t* wlo = whi + (size_t)COUT * KTOT;

    conv_wprep<<<(COUT * KTOT + 255) / 256, 256, 0, stream>>>(w, whi, wlo);

    dim3 grid(NTOT / 128, COUT / 128);   // 784 x 2
    conv_wmma<<<grid, 256, 0, stream>>>(x, whi, wlo, (float*)d_out);
}
